// Model_77747497992553
// MI455X (gfx1250) — matrix operand order repaired and checked by reading, not compiled, not run on hardware
//
#include <hip/hip_runtime.h>
#include <math.h>

typedef _Float16 v16h __attribute__((ext_vector_type(16)));
typedef _Float16 v8h  __attribute__((ext_vector_type(8)));
typedef float    v8f  __attribute__((ext_vector_type(8)));

#define TPB 128

// ---------------- fast branchless math helpers (HW transcendentals) ----------

__device__ __forceinline__ float fast_rcp(float x) {
  return __builtin_amdgcn_rcpf(x);
}
__device__ __forceinline__ float fast_rsq(float x) {
  return __builtin_amdgcn_rsqf(x);
}
// SELU, branchless: v_exp + v_cndmask, no EXEC masking between WMMAs
__device__ __forceinline__ float selu_f(float x) {
  const float S = 1.0507009873554805f;
  const float A = 1.6732632423543772f;
  float neg = S * A * (__expf(x) - 1.0f);
  return (x > 0.f) ? S * x : neg;
}
// tanh, stable + branchless: e=exp(-2|x|), t=(1-e)/(1+e), restore sign
__device__ __forceinline__ float fast_tanh(float x) {
  float e = __expf(-2.0f * fabsf(x));
  float t = (1.0f - e) * fast_rcp(1.0f + e);
  return copysignf(t, x);
}
// softplus(10x)/10, stable: max(z,0)+log(1+exp(-|z|)), z=10x
__device__ __forceinline__ float softplus10(float x) {
  float z = 10.0f * x;
  return (fmaxf(z, 0.0f) + __logf(1.0f + __expf(-fabsf(z)))) * 0.1f;
}

// ---------------- WMMA fragment loaders (CDNA5 16-bit layouts, ISA 7.12.2) ---

// A-matrix 16x32 f16: lane l holds row M=l%16.
// halves 0..7  = K = kb + (l>=16? 8:0) + 0..7      (contiguous)
// halves 8..15 = K = kb + 16 + (l>=16? 8:0) + 0..7 (contiguous)
__device__ __forceinline__ v16h load_frag_a(const _Float16* act, int rowBase,
                                            int kb, int lane) {
  int row = rowBase + (lane & 15);
  int off = (lane >= 16) ? 8 : 0;
  const _Float16* p = act + row * 64 + kb + off;
  v8h lo = *(const v8h*)(p);
  v8h hi = *(const v8h*)(p + 16);
  v16h r;
#pragma unroll
  for (int i = 0; i < 8; ++i) { r[i] = lo[i]; r[i + 8] = hi[i]; }
  return r;
}

// B-matrix 32x16 f16: lane l holds column N=l%16.
// halves 0..7 = K = kb + (l>=16?8:0) + j; halves 8..15 = K = kb + 16 + (l>=16?8:0) + (j-8)
__device__ __forceinline__ v16h load_frag_b(const _Float16* wt, int n, int kb,
                                            int lane) {
  int koff = (lane >= 16) ? 8 : 0;
  const _Float16* p = wt + n * 64 + kb + koff;
  v8h lo = *(const v8h*)(p);
  v8h hi = *(const v8h*)(p + 16);
  v16h r;
#pragma unroll
  for (int i = 0; i < 8; ++i) { r[i] = lo[i]; r[i + 8] = hi[i]; }
  return r;
}

// One hidden layer (K=64 padded, N=64): 4 N-tiles x 2 M-tiles x 2 K-steps
__device__ __forceinline__ void mlp_layer64(const _Float16* actIn,
                                            _Float16* actOut,
                                            const _Float16* wt,
                                            const float* bias, int wv,
                                            int lane) {
#pragma unroll
  for (int nt = 0; nt < 4; ++nt) {
    int n = nt * 16 + (lane & 15);
    v16h bf0 = load_frag_b(wt, n, 0, lane);
    v16h bf1 = load_frag_b(wt, n, 32, lane);
    float bv = bias[n];
#pragma unroll
    for (int mt = 0; mt < 2; ++mt) {
      int rb = wv * 32 + mt * 16;
      v16h a0 = load_frag_a(actIn, rb, 0, lane);
      v16h a1 = load_frag_a(actIn, rb, 32, lane);
      v8f acc;
#pragma unroll
      for (int r = 0; r < 8; ++r) acc[r] = bv;
      acc = __builtin_amdgcn_wmma_f32_16x16x32_f16(false, a0, false, bf0,
                                                   (short)0, acc, false, false);
      acc = __builtin_amdgcn_wmma_f32_16x16x32_f16(false, a1, false, bf1,
                                                   (short)0, acc, false, false);
      // D layout: VGPR r -> M = rb + r + (lane>=16?8:0), N = lane%16 (+nt*16)
      int rr = rb + ((lane >= 16) ? 8 : 0);
#pragma unroll
      for (int r = 0; r < 8; ++r)
        actOut[(rr + r) * 64 + n] = (_Float16)selu_f(acc[r]);
    }
  }
}

// ---------------- Kernel 1: per-vertex SH color (-> workspace) ---------------

__global__ __launch_bounds__(256) void sh_vertex_kernel(
    const float* __restrict__ vertices, const float* __restrict__ lights,
    const float* __restrict__ cam, float* __restrict__ vraw, int V) {
  int v = blockIdx.x * 256 + threadIdx.x;
  if (v >= V) return;
  float dx = vertices[v * 3 + 0] - cam[0];
  float dy = vertices[v * 3 + 1] - cam[1];
  float dz = vertices[v * 3 + 2] - cam[2];
  float inv = fast_rsq(fmaxf(dx * dx + dy * dy + dz * dz, 1e-12f));
  float x = dx * inv, y = dy * inv, z = dz * inv;
  const float C1 = 0.4886025119029199f;
  float basis[8];
  basis[0] = -C1 * y;
  basis[1] = C1 * z;
  basis[2] = -C1 * x;
  basis[3] = 1.0925484305920792f * x * y;
  basis[4] = -1.0925484305920792f * y * z;
  basis[5] = 0.31539156525252005f * (2.0f * z * z - x * x - y * y);
  basis[6] = -1.0925484305920792f * x * z;
  basis[7] = 0.5462742152960396f * (x * x - y * y);
#pragma unroll
  for (int c = 0; c < 3; ++c) {
    float s = 0.f;
#pragma unroll
    for (int k = 0; k < 8; ++k) s += basis[k] * lights[v * 24 + k * 3 + c];
    vraw[v * 3 + c] = s - 0.5f;  // vertex_color_raw includes the -0.5
  }
}

// ---------------- Kernel 2: fused per-tet pipeline + WMMA MLP ----------------

__global__ __launch_bounds__(TPB) void tet_fused_kernel(
    const float* __restrict__ vertices, const float* __restrict__ table,
    const float* __restrict__ W1, const float* __restrict__ b1,
    const float* __restrict__ W2, const float* __restrict__ b2,
    const float* __restrict__ W3, const float* __restrict__ b3,
    const int* __restrict__ indices, const float* __restrict__ vraw,
    float* __restrict__ out, int T) {
  __shared__ _Float16 sW1t[64 * 64];  // transposed [n][k], k>=40 zero
  __shared__ _Float16 sW2t[64 * 64];  // transposed [n][k]
  __shared__ _Float16 sW3t[16 * 64];  // transposed [n][k], n>=13 zero
  __shared__ _Float16 sActA[TPB * 64];
  __shared__ _Float16 sActB[TPB * 64];
  __shared__ float sO[TPB * 16];

  const int tid = threadIdx.x;
  const int lane = tid & 31;
  const int wv = tid >> 5;

  // cooperative weight staging (transposed + zero-padded, f16)
  for (int i = tid; i < 64 * 64; i += TPB) {
    int n = i >> 6, k = i & 63;
    sW1t[i] = (k < 40) ? (_Float16)W1[k * 64 + n] : (_Float16)0.f;
    sW2t[i] = (_Float16)W2[k * 64 + n];
  }
  for (int i = tid; i < 16 * 64; i += TPB) {
    int n = i >> 6, k = i & 63;
    sW3t[i] = (n < 13) ? (_Float16)W3[k * 13 + n] : (_Float16)0.f;
  }

  const int tG = blockIdx.x * TPB + tid;
  const int t = (tG < T) ? tG : (T - 1);

  // ---- gather tet vertices -------------------------------------------------
  int4 idx = ((const int4*)indices)[t];
  int iv[4] = {idx.x, idx.y, idx.z, idx.w};
  float vv[4][3];
#pragma unroll
  for (int v = 0; v < 4; ++v) {
#pragma unroll
    for (int d = 0; d < 3; ++d) vv[v][d] = vertices[iv[v] * 3 + d];
  }

  // ---- circumcenter via Cramer (A = 2*(vi-v0) + 1e-6*I) --------------------
  float e[3][3], bb[3];
#pragma unroll
  for (int i = 0; i < 3; ++i) {
    float s = 0.f;
#pragma unroll
    for (int j = 0; j < 3; ++j) {
      e[i][j] = 2.0f * (vv[i + 1][j] - vv[0][j]) + ((i == j) ? 1e-06f : 0.f);
      s += vv[i + 1][j] * vv[i + 1][j] - vv[0][j] * vv[0][j];
    }
    bb[i] = s;
  }
  float m00 = e[1][1] * e[2][2] - e[1][2] * e[2][1];
  float m01 = e[1][0] * e[2][2] - e[1][2] * e[2][0];
  float m02 = e[1][0] * e[2][1] - e[1][1] * e[2][0];
  float det = e[0][0] * m00 - e[0][1] * m01 + e[0][2] * m02;
  float invd = fast_rcp(det);
  float ccx = (bb[0] * m00 - e[0][1] * (bb[1] * e[2][2] - e[1][2] * bb[2]) +
               e[0][2] * (bb[1] * e[2][1] - e[1][1] * bb[2])) * invd;
  float ccy = (e[0][0] * (bb[1] * e[2][2] - e[1][2] * bb[2]) - bb[0] * m01 +
               e[0][2] * (e[1][0] * bb[2] - bb[1] * e[2][0])) * invd;
  float ccz = (e[0][0] * (e[1][1] * bb[2] - bb[1] * e[2][1]) -
               e[0][1] * (e[1][0] * bb[2] - bb[1] * e[2][0]) + bb[0] * m02) * invd;
  float rdx = ccx - vv[0][0], rdy = ccy - vv[0][1], rdz = ccz - vv[0][2];
  float radius = sqrtf(rdx * rdx + rdy * rdy + rdz * rdz);

  // ---- contraction (SCENE_SCALING = 1) -------------------------------------
  float nn = sqrtf(ccx * ccx + ccy * ccy + ccz * ccz);
  float sn = fmaxf(nn, 1e-12f);
  float isn = fast_rcp(sn);
  float cvs = (nn <= 1.0f) ? 1.0f : (2.0f - isn) * isn;  // cv = cvs * cc
  float crs = (nn <= 1.0f) ? 1.0f : isn * isn;
  float cr = radius * crs;
  float hx = (ccx * cvs) * 0.25f + 0.5f;  // x = (cv/2 + 1)/2
  float hy = (ccy * cvs) * 0.25f + 0.5f;
  float hz = (ccz * cvs) * 0.25f + 0.5f;

  // ---- hash-grid encoding -> f16 activation row in LDS ---------------------
  {
    _Float16* hrow = sActA + tid * 64;
#pragma unroll
    for (int lvl = 0; lvl < 10; ++lvl) {
      float res = (float)(16 << lvl);
      float xs0 = hx * res, xs1 = hy * res, xs2 = hz * res;
      float fl0 = floorf(xs0), fl1 = floorf(xs1), fl2 = floorf(xs2);
      float f0 = xs0 - fl0, f1 = xs1 - fl1, f2 = xs2 - fl2;
      unsigned xi0 = (unsigned)(int)fl0;
      unsigned xi1 = (unsigned)(int)fl1;
      unsigned xi2 = (unsigned)(int)fl2;
      float a0 = 0.f, a1 = 0.f, a2 = 0.f, a3 = 0.f;
#pragma unroll
      for (int c = 0; c < 8; ++c) {
        unsigned cx = c & 1u, cy = (c >> 1) & 1u, cz = (c >> 2) & 1u;
        unsigned h = (xi0 + cx) * 1u ^ (xi1 + cy) * 2654435761u ^
                     (xi2 + cz) * 805459861u;
        unsigned id = h & 65535u;
        float w = (cx ? f0 : 1.f - f0) * (cy ? f1 : 1.f - f1) *
                  (cz ? f2 : 1.f - f2);
        const float4 fv =
            *(const float4*)(table + ((size_t)lvl * 65536u + id) * 4);
        a0 += w * fv.x; a1 += w * fv.y; a2 += w * fv.z; a3 += w * fv.w;
      }
      float arg = 4.0f * (float)lvl * cr;  // PLS*4*n*(cr*0.5)
      float scal = erff(fast_rsq(fmaxf(arg, 1e-12f)));
      hrow[0 * 10 + lvl] = (_Float16)(a0 * scal);  // k = d*L + lvl
      hrow[1 * 10 + lvl] = (_Float16)(a1 * scal);
      hrow[2 * 10 + lvl] = (_Float16)(a2 * scal);
      hrow[3 * 10 + lvl] = (_Float16)(a3 * scal);
    }
#pragma unroll
    for (int k = 40; k < 64; ++k) hrow[k] = (_Float16)0.f;  // zero K-pad
  }
  __syncthreads();

  // ---- MLP: 40->64->64->13 on WMMA -----------------------------------------
  mlp_layer64(sActA, sActB, sW1t, b1, wv, lane);
  __syncthreads();
  mlp_layer64(sActB, sActA, sW2t, b2, wv, lane);
  __syncthreads();
  {  // layer 3: single N-tile (13 padded to 16), f32 out, no activation
    int n = lane & 15;
    v16h bf0 = load_frag_b(sW3t, n, 0, lane);
    v16h bf1 = load_frag_b(sW3t, n, 32, lane);
    float bv = (n < 13) ? b3[n] : 0.f;
#pragma unroll
    for (int mt = 0; mt < 2; ++mt) {
      int rb = wv * 32 + mt * 16;
      v16h fa0 = load_frag_a(sActA, rb, 0, lane);
      v16h fa1 = load_frag_a(sActA, rb, 32, lane);
      v8f acc;
#pragma unroll
      for (int r = 0; r < 8; ++r) acc[r] = bv;
      acc = __builtin_amdgcn_wmma_f32_16x16x32_f16(false, fa0, false, bf0,
                                                   (short)0, acc, false, false);
      acc = __builtin_amdgcn_wmma_f32_16x16x32_f16(false, fa1, false, bf1,
                                                   (short)0, acc, false, false);
      int rr = rb + ((lane >= 16) ? 8 : 0);
#pragma unroll
      for (int r = 0; r < 8; ++r) sO[(rr + r) * 16 + n] = acc[r];
    }
  }
  __syncthreads();

  // ---- epilogue (all branchless HW transcendentals) ------------------------
  float o[13];
#pragma unroll
  for (int j = 0; j < 13; ++j) o[j] = sO[tid * 16 + j];

  float density = __expf(fminf(fmaxf(o[0] - 1.0f, -15.0f), 15.0f));
  float basec[3], g[3][3];
#pragma unroll
  for (int c = 0; c < 3; ++c) {
    basec[c] = o[1 + c] + 0.5f;
    float bm = fmaxf(basec[c], 0.0f);
#pragma unroll
    for (int d = 0; d < 3; ++d)
      g[c][d] = bm * fast_tanh(0.1f * o[4 + c * 3 + d]);
  }

  float res16[16];
  res16[0] = ccx; res16[1] = ccy; res16[2] = ccz; res16[3] = density;
#pragma unroll
  for (int v = 0; v < 4; ++v) {
    float ox = vv[v][0] - ccx, oy = vv[v][1] - ccy, oz = vv[v][2] - ccz;
    float oinv = fast_rsq(fmaxf(ox * ox + oy * oy + oz * oz, 1e-12f));
    ox *= oinv; oy *= oinv; oz *= oinv;
#pragma unroll
    for (int c = 0; c < 3; ++c) {
      float gc = g[c][0] * ox + g[c][1] * oy + g[c][2] * oz;
      float val = vraw[iv[v] * 3 + c] + basec[c] + gc;
      res16[4 + v * 3 + c] = softplus10(val);
    }
  }

  if (tG < T) {
    float4* po = (float4*)(out + (size_t)tG * 16);
    po[0] = make_float4(res16[0], res16[1], res16[2], res16[3]);
    po[1] = make_float4(res16[4], res16[5], res16[6], res16[7]);
    po[2] = make_float4(res16[8], res16[9], res16[10], res16[11]);
    po[3] = make_float4(res16[12], res16[13], res16[14], res16[15]);
  }
}

// ---------------- host-side launcher -----------------------------------------

extern "C" void kernel_launch(void* const* d_in, const int* in_sizes, int n_in,
                              void* d_out, int out_size, void* d_ws,
                              size_t ws_size, hipStream_t stream) {
  const float* vertices = (const float*)d_in[0];
  const float* lights   = (const float*)d_in[1];
  const float* cam      = (const float*)d_in[2];
  const float* table    = (const float*)d_in[3];
  const float* W1       = (const float*)d_in[4];
  const float* b1       = (const float*)d_in[5];
  const float* W2       = (const float*)d_in[6];
  const float* b2       = (const float*)d_in[7];
  const float* W3       = (const float*)d_in[8];
  const float* b3       = (const float*)d_in[9];
  const int*   indices  = (const int*)d_in[10];

  const int V = in_sizes[0] / 3;
  const int T = in_sizes[10] / 4;

  float* vraw = (float*)d_ws;  // V*3 floats of scratch
  float* out  = (float*)d_out;

  sh_vertex_kernel<<<(V + 255) / 256, 256, 0, stream>>>(vertices, lights, cam,
                                                        vraw, V);
  tet_fused_kernel<<<(T + TPB - 1) / TPB, TPB, 0, stream>>>(
      vertices, table, W1, b1, W2, b2, W3, b3, indices, vraw, out, T);
}
